// MultiHeadSelfAttention_82970178224305
// MI455X (gfx1250) — matrix operand order repaired and checked by reading, not compiled, not run on hardware
//
#include <hip/hip_runtime.h>

// ---------------------------------------------------------------------------
// Types / WMMA fragment helpers (CDNA5 gfx1250, wave32)
// ---------------------------------------------------------------------------
typedef __bf16 bf16;
typedef __attribute__((ext_vector_type(16))) __bf16 v16bf;
typedef __attribute__((ext_vector_type(8)))  float  v8f;
typedef __attribute__((ext_vector_type(4)))  float  v4f;

union Frag32B { v16bf v; v4f f[2]; };

// A-matrix 16x32 bf16 fragment: lane<16 row=lane, k = {0..7,16..23};
// lane>=16 row=lane-16, k = {8..15,24..31}.  (ISA 7.12.2)
__device__ __forceinline__ v16bf wmma_a_frag(const bf16* __restrict__ row, int kb, int hi) {
  const bf16* p = row + kb + (hi ? 8 : 0);
  Frag32B f;
  f.f[0] = *(const v4f*)p;
  f.f[1] = *(const v4f*)(p + 16);
  return f.v;
}
// B-matrix 32x16 bf16 fragment: lane<16 col=lane, k = {0..7,16..23}; lane>=16 col=lane-16, k = {8..15,24..31}.
__device__ __forceinline__ v16bf wmma_b_frag(const bf16* __restrict__ row, int kb, int hi) {
  const bf16* p = row + kb + (hi ? 8 : 0);
  Frag32B f;
  f.f[0] = *(const v4f*)p;
  f.f[1] = *(const v4f*)(p + 16);
  return f.v;
}
__device__ __forceinline__ v8f wmma_bf(v16bf a, v16bf b, v8f c) {
  return __builtin_amdgcn_wmma_f32_16x16x32_bf16(false, a, false, b, (short)0, c,
                                                 false, false);
}

// Problem constants
#define SEQ   1024
#define DIM   1024
#define NB    4
#define NH    16
#define DH    64
#define MROWS 4096        // NB*SEQ
#define JP    288         // padded relpos dim (257 -> 288, mult of 32)

// ---------------------------------------------------------------------------
// Packing kernels
// ---------------------------------------------------------------------------
__global__ void cvt_f32_bf16(const float* __restrict__ in, bf16* __restrict__ out, int n) {
  int i = blockIdx.x * blockDim.x + threadIdx.x;
  if (i < n) out[i] = (bf16)in[i];
}

// WT[n][k] = W[k][n]; read W coalesced along n.
__global__ void transpose_w_bf16(const float* __restrict__ W, bf16* __restrict__ WT) {
  int i = blockIdx.x * blockDim.x + threadIdx.x;
  if (i < DIM * DIM) {
    int k = i / DIM, n = i % DIM;
    WT[n * DIM + k] = (bf16)W[k * DIM + n];
  }
}

// posKpad: [272][64] bf16, rows >=257 zero
__global__ void pack_posK(const float* __restrict__ posK, bf16* __restrict__ out) {
  int i = blockIdx.x * blockDim.x + threadIdx.x;
  if (i < 272 * 64) {
    int j = i >> 6;
    out[i] = (j < 257) ? (bf16)posK[i] : (bf16)0.0f;
  }
}

// posVTpad: [64][288] bf16, posVT[d][j] = posV[j][d], cols >=257 zero
__global__ void pack_posVT(const float* __restrict__ posV, bf16* __restrict__ out) {
  int i = blockIdx.x * blockDim.x + threadIdx.x;
  if (i < 64 * JP) {
    int dd = i / JP, j = i % JP;
    out[i] = (j < 257) ? (bf16)posV[j * 64 + dd] : (bf16)0.0f;
  }
}

// ---------------------------------------------------------------------------
// Generic bf16 WMMA GEMM:  C[M=4096, N=1024] = A @ Bt^T + bias
//   A  : [4096][1024] bf16 row-major
//   Bt : [1024][1024] bf16, Bt[n][k] = B[k][n]
// mode 0: out bf16 [B][H][L][64]   (Q, K)
// mode 1: out bf16 [B][H][64][L]   (V transposed)
// mode 2: out fp32 [4096][1024]    (final projection)
// Block = 256 threads = 8 waves; wave tile 32x32; WG tile 128(M) x 64(N).
// ---------------------------------------------------------------------------
__global__ __launch_bounds__(256) void gemm_bf16(
    const bf16* __restrict__ A, const bf16* __restrict__ Bt,
    const float* __restrict__ bias, void* __restrict__ out, int mode)
{
  const int lane = threadIdx.x & 31;
  const int wid  = threadIdx.x >> 5;
  const int lo = lane & 15, hi = lane >> 4;
  const int wm = wid & 3, wn = wid >> 2;
  const int m0 = blockIdx.x * 128 + wm * 32;
  const int n0 = blockIdx.y * 64  + wn * 32;

  v8f acc[2][2] = {};
  const bf16* ar0 = A  + (m0 + lo)      * DIM;
  const bf16* ar1 = A  + (m0 + 16 + lo) * DIM;
  const bf16* br0 = Bt + (n0 + lo)      * DIM;
  const bf16* br1 = Bt + (n0 + 16 + lo) * DIM;

  for (int kb = 0; kb < DIM; kb += 32) {
    v16bf a0 = wmma_a_frag(ar0, kb, hi);
    v16bf a1 = wmma_a_frag(ar1, kb, hi);
    v16bf b0 = wmma_b_frag(br0, kb, hi);
    v16bf b1 = wmma_b_frag(br1, kb, hi);
    acc[0][0] = wmma_bf(a0, b0, acc[0][0]);
    acc[0][1] = wmma_bf(a0, b1, acc[0][1]);
    acc[1][0] = wmma_bf(a1, b0, acc[1][0]);
    acc[1][1] = wmma_bf(a1, b1, acc[1][1]);
  }

#pragma unroll
  for (int tm = 0; tm < 2; tm++) {
#pragma unroll
    for (int tn = 0; tn < 2; tn++) {
      int ng = n0 + tn * 16 + lo;
      float bv = bias[ng];
#pragma unroll
      for (int i = 0; i < 8; i++) {
        int mg = m0 + tm * 16 + i + hi * 8;
        float val = acc[tm][tn][i] + bv;
        if (mode == 2) {
          ((float*)out)[mg * DIM + ng] = val;
        } else {
          int b = mg >> 10, l = mg & 1023;
          int h = ng >> 6,  dd = ng & 63;
          if (mode == 0)
            ((bf16*)out)[(((b * NH + h) * SEQ) + l) * DH + dd] = (bf16)val;
          else
            ((bf16*)out)[(((b * NH + h) * DH) + dd) * SEQ + l] = (bf16)val;
        }
      }
    }
  }
}

// ---------------------------------------------------------------------------
// Attention kernel: one wave per (b, h, 16-row q-tile).
//   Q,K : [B][H][L][64] bf16       Vt : [B][H][64][L] bf16
//   posKp : [272][64] bf16          posVTp : [64][288] bf16
//   outA : [B*L][1024] bf16  (attn output, pre-Wo)
// Two-pass softmax; S2 via P-table gather; O2 via relpos bucket GEMM.
// ---------------------------------------------------------------------------
__global__ __launch_bounds__(32) void attn_kernel(
    const bf16* __restrict__ Q, const bf16* __restrict__ Km,
    const bf16* __restrict__ Vt,
    const bf16* __restrict__ posKp, const bf16* __restrict__ posVTp,
    const float* __restrict__ pad_mask, bf16* __restrict__ outA)
{
  __shared__ float Pl[16][JP];   // P[q_local][j] = Q . posK
  __shared__ float Cl[16][JP];   // relpos bucket weights
  __shared__ float Zr[16];       // 1/rowsum
  __shared__ __align__(16) bf16 stage[16][32]; // C-layout -> A-layout transpose
  __shared__ float pm[SEQ];

  const int lane = threadIdx.x;
  const int lo = lane & 15, hi = lane >> 4;
  const int q0 = blockIdx.x * 16;
  const int h  = blockIdx.y;
  const int b  = blockIdx.z;
  const int bh = b * NH + h;

  const bf16* Qb = Q  + (size_t)bh * SEQ * DH;
  const bf16* Kb = Km + (size_t)bh * SEQ * DH;
  const bf16* Vb = Vt + (size_t)bh * DH * SEQ;

  for (int i = lane; i < SEQ; i += 32) pm[i] = pad_mask[b * SEQ + i];
  for (int i = lane; i < 16 * JP; i += 32) (&Cl[0][0])[i] = 0.0f;

  // Q fragments (reused everywhere): d-chunks [0,32) and [32,64)
  const bf16* qrow = Qb + (q0 + lo) * DH;
  v16bf qa0 = wmma_a_frag(qrow, 0, hi);
  v16bf qa1 = wmma_a_frag(qrow, 32, hi);

  // ---- Phase P: P[16][272] = Q_tile @ posK^T (17 WMMA n-tiles) ----
  for (int jt = 0; jt < 17; jt++) {
    const bf16* prow = posKp + (jt * 16 + lo) * DH;
    v16bf pb0 = wmma_b_frag(prow, 0, hi);
    v16bf pb1 = wmma_b_frag(prow, 32, hi);
    v8f c = {};
    c = wmma_bf(qa0, pb0, c);
    c = wmma_bf(qa1, pb1, c);
#pragma unroll
    for (int i = 0; i < 8; i++) Pl[i + hi * 8][jt * 16 + lo] = c[i];
  }
  __syncthreads();

  const float scale = 0.125f; // 1/sqrt(64)

  // ---- Phase A: row max ----
  float lm[8];
#pragma unroll
  for (int i = 0; i < 8; i++) lm[i] = -1e30f;
  for (int kb = 0; kb < SEQ; kb += 16) {
    const bf16* krow = Kb + (kb + lo) * DH;
    v8f c = {};
    c = wmma_bf(qa0, wmma_b_frag(krow, 0, hi), c);
    c = wmma_bf(qa1, wmma_b_frag(krow, 32, hi), c);
    int kk = kb + lo;
    float mask = pm[kk];
#pragma unroll
    for (int i = 0; i < 8; i++) {
      int m = i + hi * 8;
      int diff = kk - (q0 + m);
      if (diff < -128) diff = -128;
      if (diff > 128) diff = 128;
      float s = (c[i] + Pl[m][diff + 128]) * scale * mask;
      lm[i] = fmaxf(lm[i], s);
    }
  }
#pragma unroll
  for (int d = 1; d < 16; d <<= 1)
#pragma unroll
    for (int i = 0; i < 8; i++) lm[i] = fmaxf(lm[i], __shfl_xor(lm[i], d, 32));

  // ---- Phase B: p = exp(S - max); rowsum; O1 += p @ V; buckets C[m][j] += p ----
  float ls[8] = {0.f, 0.f, 0.f, 0.f, 0.f, 0.f, 0.f, 0.f};
  v8f o[4] = {};
  for (int kb = 0; kb < SEQ; kb += 32) {
#pragma unroll
    for (int sub = 0; sub < 2; sub++) {
      int kbs = kb + sub * 16;
      const bf16* krow = Kb + (kbs + lo) * DH;
      v8f c = {};
      c = wmma_bf(qa0, wmma_b_frag(krow, 0, hi), c);
      c = wmma_bf(qa1, wmma_b_frag(krow, 32, hi), c);
      int kk = kbs + lo;
      float mask = pm[kk];
#pragma unroll
      for (int i = 0; i < 8; i++) {
        int m = i + hi * 8;
        int diff = kk - (q0 + m);
        if (diff < -128) diff = -128;
        if (diff > 128) diff = 128;
        float s = (c[i] + Pl[m][diff + 128]) * scale * mask;
        float p = __expf(s - lm[i]);
        ls[i] += p;
        atomicAdd(&Cl[m][diff + 128], p);
        stage[m][sub * 16 + lo] = (bf16)p;
      }
    }
    __syncthreads(); // single-wave WG: S_NOP + compiler LDS fence
    // A-fragment of probabilities from staging (C-layout -> A-layout)
    Frag32B af;
    const bf16* sp = &stage[lo][hi ? 8 : 0];
    af.f[0] = *(const v4f*)sp;
    af.f[1] = *(const v4f*)(sp + 16);
#pragma unroll
    for (int dt = 0; dt < 4; dt++) {
      const bf16* vrow = Vb + (dt * 16 + lo) * SEQ;
      o[dt] = wmma_bf(af.v, wmma_b_frag(vrow, kb, hi), o[dt]);
    }
    __syncthreads();
  }

  // row sums -> reciprocal; normalize O accumulators
#pragma unroll
  for (int d = 1; d < 16; d <<= 1)
#pragma unroll
    for (int i = 0; i < 8; i++) ls[i] += __shfl_xor(ls[i], d, 32);
  float rz[8];
#pragma unroll
  for (int i = 0; i < 8; i++) rz[i] = 1.0f / ls[i];
#pragma unroll
  for (int dt = 0; dt < 4; dt++)
#pragma unroll
    for (int i = 0; i < 8; i++) o[dt][i] *= rz[i];
  if (lo == 0)
#pragma unroll
    for (int i = 0; i < 8; i++) Zr[i + hi * 8] = rz[i];
  __syncthreads();

  // normalize bucket weights
  for (int idx = lane; idx < 16 * JP; idx += 32) {
    int m = idx / JP;
    (&Cl[0][0])[idx] *= Zr[m];
  }
  __syncthreads();

  // ---- Phase C: O += Cbuckets[16][288] @ posV[288][64] ----
  for (int jb = 0; jb < JP; jb += 32) {
    const float* crow = &Cl[lo][jb + (hi ? 8 : 0)];
    v16bf cf = {};
#pragma unroll
    for (int e = 0; e < 8; e++) cf[e] = (bf16)crow[e];
#pragma unroll
    for (int e = 0; e < 8; e++) cf[8 + e] = (bf16)crow[16 + e];
#pragma unroll
    for (int dt = 0; dt < 4; dt++) {
      const bf16* vrow = posVTp + (dt * 16 + lo) * JP;
      o[dt] = wmma_bf(cf, wmma_b_frag(vrow, jb, hi), o[dt]);
    }
  }

  // ---- store attn output [B*L][1024] bf16 ----
#pragma unroll
  for (int dt = 0; dt < 4; dt++) {
    int dd = h * DH + dt * 16 + lo;
#pragma unroll
    for (int i = 0; i < 8; i++) {
      int q = q0 + i + hi * 8;
      outA[(size_t)(b * SEQ + q) * DIM + dd] = (bf16)o[dt][i];
    }
  }
}

// ---------------------------------------------------------------------------
// Launcher
// ---------------------------------------------------------------------------
extern "C" void kernel_launch(void* const* d_in, const int* in_sizes, int n_in,
                              void* d_out, int out_size, void* d_ws, size_t ws_size,
                              hipStream_t stream) {
  const float* x    = (const float*)d_in[0];
  const float* pad  = (const float*)d_in[1];
  const float* Wq   = (const float*)d_in[2];
  const float* bq   = (const float*)d_in[3];
  const float* Wk   = (const float*)d_in[4];
  const float* bk   = (const float*)d_in[5];
  const float* Wv   = (const float*)d_in[6];
  const float* bv   = (const float*)d_in[7];
  const float* Wo   = (const float*)d_in[8];
  const float* bo   = (const float*)d_in[9];
  const float* posK = (const float*)d_in[10];
  const float* posV = (const float*)d_in[11];

  char* ws = (char*)d_ws;
  size_t off = 0;
  bf16* xb   = (bf16*)(ws + off); off += (size_t)MROWS * DIM * 2;   // 8 MiB
  bf16* WqT  = (bf16*)(ws + off); off += (size_t)DIM * DIM * 2;
  bf16* WkT  = (bf16*)(ws + off); off += (size_t)DIM * DIM * 2;
  bf16* WvT  = (bf16*)(ws + off); off += (size_t)DIM * DIM * 2;
  bf16* WoT  = (bf16*)(ws + off); off += (size_t)DIM * DIM * 2;
  bf16* Qw   = (bf16*)(ws + off); off += (size_t)MROWS * DIM * 2;
  bf16* Kw   = (bf16*)(ws + off); off += (size_t)MROWS * DIM * 2;
  bf16* Vtw  = (bf16*)(ws + off); off += (size_t)MROWS * DIM * 2;
  bf16* attn = (bf16*)(ws + off); off += (size_t)MROWS * DIM * 2;
  bf16* posKp  = (bf16*)(ws + off); off += 272 * 64 * 2;
  bf16* posVTp = (bf16*)(ws + off); off += 64 * JP * 2;

  // ---- pack / convert ----
  cvt_f32_bf16<<<(MROWS * DIM + 255) / 256, 256, 0, stream>>>(x, xb, MROWS * DIM);
  transpose_w_bf16<<<(DIM * DIM + 255) / 256, 256, 0, stream>>>(Wq, WqT);
  transpose_w_bf16<<<(DIM * DIM + 255) / 256, 256, 0, stream>>>(Wk, WkT);
  transpose_w_bf16<<<(DIM * DIM + 255) / 256, 256, 0, stream>>>(Wv, WvT);
  transpose_w_bf16<<<(DIM * DIM + 255) / 256, 256, 0, stream>>>(Wo, WoT);
  pack_posK<<<(272 * 64 + 255) / 256, 256, 0, stream>>>(posK, posKp);
  pack_posVT<<<(64 * JP + 255) / 256, 256, 0, stream>>>(posV, posVTp);

  // ---- Q/K/V projections (WMMA) ----
  dim3 gg(MROWS / 128, DIM / 64);
  gemm_bf16<<<gg, 256, 0, stream>>>(xb, WqT, bq, Qw, 0);
  gemm_bf16<<<gg, 256, 0, stream>>>(xb, WkT, bk, Kw, 0);
  gemm_bf16<<<gg, 256, 0, stream>>>(xb, WvT, bv, Vtw, 1);

  // ---- attention (WMMA) ----
  dim3 ga(SEQ / 16, NH, NB);
  attn_kernel<<<ga, 32, 0, stream>>>(Qw, Kw, Vtw, posKp, posVTp, pad, attn);

  // ---- output projection (WMMA, fp32 out) ----
  gemm_bf16<<<gg, 256, 0, stream>>>(attn, WoT, bo, d_out, 2);
}
